// ArcLoss_74191265071115
// MI455X (gfx1250) — matrix operand order repaired and checked by reading, not compiled, not run on hardware
//
#include <hip/hip_runtime.h>
#include <math.h>

typedef __attribute__((ext_vector_type(16))) __bf16 v16bf;
typedef __attribute__((ext_vector_type(8)))  __bf16 v8bf;
typedef __attribute__((ext_vector_type(8)))  float  v8f;
typedef unsigned int u32x4 __attribute__((ext_vector_type(4)));
typedef int          i32x4 __attribute__((ext_vector_type(4)));
typedef int          i32x8 __attribute__((ext_vector_type(8)));

#define KD      512        // feature dim D
#define TILE_M  128
#define TILE_N  128
#define KSTEP   32         // K per WMMA
#define NKSTEP  (KD / KSTEP)
#define B_ROWS  1024
#define C_CLS   100000

static constexpr float ARC_M  = 0.5f;
static constexpr float ARC_S  = 64.0f;

// ---------------------------------------------------------------------------
// L2-normalize rows of a [rows, 512] f32 matrix into bf16.
// ---------------------------------------------------------------------------
__global__ void __launch_bounds__(256)
norm_rows_to_bf16(const float* __restrict__ in, __bf16* __restrict__ out) {
    const int row = blockIdx.x;
    const int t   = threadIdx.x;
    const float* r = in + (size_t)row * KD;

    float2 v = *(const float2*)(r + t * 2);
    float ss = v.x * v.x + v.y * v.y;

    __shared__ float red[256];
    red[t] = ss;
    __syncthreads();
    #pragma unroll
    for (int s = 128; s > 0; s >>= 1) {
        if (t < s) red[t] += red[t + s];
        __syncthreads();
    }
    const float norm  = sqrtf(red[0]);
    const float scale = 1.0f / fmaxf(norm, 1e-12f);

    __bf16* o = out + (size_t)row * KD;
    o[t * 2 + 0] = (__bf16)(v.x * scale);
    o[t * 2 + 1] = (__bf16)(v.y * scale);
}

// ---------------------------------------------------------------------------
// Tensor Data Mover: DMA a 128x32 bf16 tile (row stride 512 elems) into LDS.
// D# built per CDNA5 ISA 8.3/8.4 (2-D tensor, groups 2/3 NULL->zeros).
// ---------------------------------------------------------------------------
#if __has_builtin(__builtin_amdgcn_tensor_load_to_lds)
#define HAVE_TDM 1
__device__ __forceinline__ void tdm_load_a_tile(unsigned ldsAddr,
                                                unsigned long long gaddr) {
    u32x4 g0;
    g0[0] = 1u;                                      // count=1, user mode
    g0[1] = ldsAddr;                                 // lds_addr (bytes)
    g0[2] = (unsigned)(gaddr & 0xffffffffu);         // global_addr[31:0]
    g0[3] = (unsigned)((gaddr >> 32) & 0x01ffffffu)  // global_addr[56:32]
          | (2u << 30);                              // type = 2 ("image")
    i32x8 g1;
    g1[0] = 0x00010000;            // data_size=1 (2B), no multicast/flags
    g1[1] = (int)(512u << 16);     // tensor_dim0 = 512 (bits 79:48, low half)
    g1[2] = (int)(128u << 16);     // tensor_dim0 hi=0 | tensor_dim1 = 128
    g1[3] = (int)(32u  << 16);     // tensor_dim1 hi=0 | tile_dim0 = 32
    g1[4] = 128;                   // tile_dim1 = 128, tile_dim2 = 0
    g1[5] = 512;                   // tensor_dim0_stride = 512 (48b, low dword)
    g1[6] = 0;
    g1[7] = 0;
    i32x4 z4 = {0, 0, 0, 0};
#if __clang_major__ >= 23
    i32x8 z8 = {0, 0, 0, 0, 0, 0, 0, 0};
    __builtin_amdgcn_tensor_load_to_lds(g0, g1, z4, z4, z8, 0);
#else
    __builtin_amdgcn_tensor_load_to_lds(g0, g1, z4, z4, 0);
#endif
}
#else
#define HAVE_TDM 0
#endif

// ---------------------------------------------------------------------------
// out[b, c] = S * dot(xn[b], wn[c]) via v_wmma_f32_16x16x32_bf16.
// Block = 256 threads = 8 waves; tile 128(M) x 128(N); wave patch 64x32.
// A tiles double-buffered in LDS, streamed by the TDM (TENSORcnt) while
// the waves run WMMA on the other buffer; B streams from L2-resident bf16.
// ---------------------------------------------------------------------------
__global__ void __launch_bounds__(256)
arcface_gemm_bf16(const __bf16* __restrict__ xn,
                  const __bf16* __restrict__ wn,
                  float* __restrict__ out) {
    __shared__ __bf16 sA[2][TILE_M * KSTEP];   // 2 x 8 KB

    const int mBase = blockIdx.y * TILE_M;
    const int nBase = blockIdx.x * TILE_N;
    const int tid   = threadIdx.x;
    const int lane  = tid & 31;
    const int wm    = (tid >> 5) >> 2;   // 0..1  -> M offset wm*64
    const int wnq   = (tid >> 5) & 3;    // 0..3  -> N offset wnq*32
    const int lrow  = lane & 15;
    const bool hiHalf = (lane >= 16);

    v8f acc[4][2];
    #pragma unroll
    for (int m = 0; m < 4; ++m)
        #pragma unroll
        for (int n = 0; n < 2; ++n)
            acc[m][n] = {};

    const int kofsA    = hiHalf ? 16 : 0;   // A-layout per-lane-half K split
    const int kofsBrel = hiHalf ? 16 : 0;   // B-layout per-lane-half K split

    // B row pointers (clamped for the ragged last N block).
    const __bf16* bPtr[2];
    #pragma unroll
    for (int n = 0; n < 2; ++n) {
        int col = nBase + wnq * 32 + n * 16 + lrow;
        col = col < C_CLS ? col : C_CLS - 1;
        bPtr[n] = wn + (size_t)col * KD + kofsBrel;
    }

#if HAVE_TDM
    // Flat->LDS aperture maps addr[31:0], so low 32 bits = LDS byte offset.
    const unsigned ldsA[2] = { (unsigned)(uintptr_t)&sA[0][0],
                               (unsigned)(uintptr_t)&sA[1][0] };
    const unsigned long long aBase =
        (unsigned long long)(uintptr_t)(xn + (size_t)mBase * KD);

    if (tid < 32) tdm_load_a_tile(ldsA[0], aBase);   // prologue: tile 0

    for (int kk = 0; kk < NKSTEP; ++kk) {
        const int cur = kk & 1;
        if (tid < 32) __builtin_amdgcn_s_wait_tensorcnt(0);
        __syncthreads();   // tile `cur` visible to all waves;
                           // also proves buffer cur^1 is no longer being read
        if (tid < 32 && kk + 1 < NKSTEP)
            tdm_load_a_tile(ldsA[cur ^ 1],
                            aBase + (unsigned long long)(kk + 1) * KSTEP * 2);

        const __bf16* sAcur = sA[cur];
        const int k0 = kk * KSTEP;
#else
    // Fallback: cooperative staging (each thread moves 32 B).
    const int aRow = tid >> 1;
    const int aCol = (tid & 1) * 16;
    const __bf16* aGlobRow = xn + (size_t)(mBase + aRow) * KD + aCol;
    for (int kk = 0; kk < NKSTEP; ++kk) {
        const int k0 = kk * KSTEP;
        *(v16bf*)&sA[0][aRow * KSTEP + aCol] = *(const v16bf*)(aGlobRow + k0);
        __syncthreads();
        const __bf16* sAcur = sA[0];
#endif

        // B fragments: 16 contiguous bf16 per lane (k = 16h + s, same order as A).
        v16bf bFrag[2];
        #pragma unroll
        for (int n = 0; n < 2; ++n)
            bFrag[n] = *(const v16bf*)(bPtr[n] + k0);

        // A fragments from LDS (k = 16h + s, matching B) + WMMA.
        #pragma unroll
        for (int m = 0; m < 4; ++m) {
            const __bf16* ap = &sAcur[(wm * 64 + m * 16 + lrow) * KSTEP + kofsA];
            union { v16bf v; v8bf h[2]; } ua;
            ua.h[0] = *(const v8bf*)(ap);
            ua.h[1] = *(const v8bf*)(ap + 8);
            #pragma unroll
            for (int n = 0; n < 2; ++n) {
                acc[m][n] = __builtin_amdgcn_wmma_f32_16x16x32_bf16(
                    /*neg_a=*/false, ua.v,
                    /*neg_b=*/false, bFrag[n],
                    /*c_mod=*/(short)0, acc[m][n],
                    /*reuse_a=*/false, /*reuse_b=*/false);
            }
        }
#if !HAVE_TDM
        __syncthreads();
#endif
    }

    // Epilogue: C/D layout -> VGPR r holds M=r (lanes 0-15) / M=r+8 (lanes 16-31).
    const int rowAdd = hiHalf ? 8 : 0;
    #pragma unroll
    for (int m = 0; m < 4; ++m) {
        const int rowBase = mBase + wm * 64 + m * 16 + rowAdd;
        #pragma unroll
        for (int n = 0; n < 2; ++n) {
            const int col = nBase + wnq * 32 + n * 16 + lrow;
            if (col < C_CLS) {
                #pragma unroll
                for (int r = 0; r < 8; ++r) {
                    out[(size_t)(rowBase + r) * C_CLS + col] = acc[m][n][r] * ARC_S;
                }
            }
        }
    }
}

// ---------------------------------------------------------------------------
// ArcFace margin at (b, target[b]); out already holds S*cos.
// ---------------------------------------------------------------------------
__global__ void margin_fixup(const int* __restrict__ target,
                             float* __restrict__ out) {
    const int b = blockIdx.x * blockDim.x + threadIdx.x;
    if (b >= B_ROWS) return;
    const int t = target[b];
    const size_t idx = (size_t)b * C_CLS + t;
    const float cost = out[idx] * (1.0f / ARC_S);
    if (cost > 0.0f) {                 // easy_margin keep-branch otherwise
        const float c  = fminf(fmaxf(cost, -1.0f), 1.0f);
        const float nz = cosf(acosf(c) + ARC_M);
        out[idx] = nz * ARC_S;
    }
}

// ---------------------------------------------------------------------------
extern "C" void kernel_launch(void* const* d_in, const int* in_sizes, int n_in,
                              void* d_out, int out_size, void* d_ws, size_t ws_size,
                              hipStream_t stream) {
    (void)in_sizes; (void)n_in; (void)out_size; (void)ws_size;

    const float* x      = (const float*)d_in[0];   // [1024, 512] f32
    const float* weight = (const float*)d_in[1];   // [100000, 512] f32
    const int*   target = (const int*)d_in[2];     // [1024]
    float*       out    = (float*)d_out;           // [1024, 100000] f32

    // Workspace: bf16 wn (102.4 MB, fits 192 MB L2) then bf16 xn (1 MB).
    __bf16* wn = (__bf16*)d_ws;
    __bf16* xn = (__bf16*)((char*)d_ws + (size_t)C_CLS * KD * sizeof(__bf16));

    norm_rows_to_bf16<<<C_CLS, 256, 0, stream>>>(weight, wn);
    norm_rows_to_bf16<<<B_ROWS, 256, 0, stream>>>(x, xn);

    dim3 grid((C_CLS + TILE_N - 1) / TILE_N, B_ROWS / TILE_M);
    arcface_gemm_bf16<<<grid, 256, 0, stream>>>(xn, wn, out);

    margin_fixup<<<(B_ROWS + 255) / 256, 256, 0, stream>>>(target, out);
}
